// DiTSwinTransformerBlock_87582973100529
// MI455X (gfx1250) — matrix operand order repaired and checked by reading, not compiled, not run on hardware
//
#include <hip/hip_runtime.h>
#include <hip/hip_bf16.h>

typedef __bf16 bf16;
typedef __attribute__((ext_vector_type(16))) __bf16 v16bf;
typedef __attribute__((ext_vector_type(8)))  __bf16 v8bf;
typedef __attribute__((ext_vector_type(8)))  float  v8f;

#define WAVE ((int)(threadIdx.x >> 5))
#define LANE ((int)(threadIdx.x & 31))

// ---------------------------------------------------------------------------
// WMMA fragment helpers (CDNA5 16x16x32 bf16, wave32, contiguous k = 16*(lane/16) + s)
// ---------------------------------------------------------------------------
__device__ __forceinline__ v16bf make_frag(v8bf lo, v8bf hi) {
  v16bf f;
#pragma unroll
  for (int i = 0; i < 8; ++i) { f[i] = lo[i]; f[i + 8] = hi[i]; }
  return f;
}

// B fragment: 32x16 bf16 tile, B supplied TRANSPOSED as Bt[N][K] so each lane
// reads a contiguous run of 16 K-values for its column n = n0 + (lane&15).
__device__ __forceinline__ v16bf load_b_frag(const bf16* Bt, int ldk, int n0, int k0) {
  const bf16* p = Bt + (size_t)(n0 + (LANE & 15)) * ldk + k0 + ((LANE >> 4) << 4);
  return make_frag(*(const v8bf*)p, *(const v8bf*)(p + 8));
}

__device__ __forceinline__ v8f wmma_bf16(v16bf a, v16bf b, v8f c) {
  return __builtin_amdgcn_wmma_f32_16x16x32_bf16(false, a, false, b, (short)0, c,
                                                 false, false);
}

// ---------------------------------------------------------------------------
// Prep kernels
// ---------------------------------------------------------------------------
__global__ __launch_bounds__(256) void k_transpose_bf16(const float* W, bf16* Wt,
                                                        int K, int N) {
  long i = (long)blockIdx.x * 256 + threadIdx.x;
  if (i >= (long)K * N) return;
  int k = (int)(i / N), n = (int)(i % N);
  Wt[(size_t)n * K + k] = (bf16)W[i];
}

// mod = cond(16x128) @ W(128x512) + b  -> (16x512)
__global__ __launch_bounds__(256) void k_mod(const float* cond, const float* W,
                                             const float* b, float* mod) {
  int j = blockIdx.x * 256 + threadIdx.x;
  if (j >= 16 * 512) return;
  int bi = j >> 9, c = j & 511;
  float s = b[c];
  for (int k = 0; k < 128; ++k) s += cond[bi * 128 + k] * W[k * 512 + c];
  mod[j] = s;
}

// bias[h][i][j] from relative-position table (WS=8 -> 15x15 table, 8 heads)
__global__ __launch_bounds__(256) void k_bias(const float* rpb, float* bm) {
  int i = blockIdx.x * 256 + threadIdx.x;
  if (i >= 8 * 64 * 64) return;
  int h = i >> 12, rem = i & 4095, a = rem >> 6, b2 = rem & 63;
  int idx = ((a >> 3) - (b2 >> 3) + 7) * 15 + ((a & 7) - (b2 & 7) + 7);
  bm[i] = rpb[idx * 8 + h];
}

// ---------------------------------------------------------------------------
// adaLN (one wave32 per token) + optional shifted window partition
// ---------------------------------------------------------------------------
__global__ __launch_bounds__(256) void k_adaln(const float* xin, const float* mod,
                                               const float* gamma, const float* beta,
                                               bf16* out, int windowed) {
  long wr = (long)blockIdx.x * 8 + WAVE;
  int lane = LANE;
  size_t tok; int b;
  if (windowed) {
    int w = (int)(wr >> 6), tn = (int)(wr & 63);
    b = w >> 6;
    int wy = (w >> 3) & 7, wx = w & 7;
    int y  = ((wy << 3) + (tn >> 3) + 4) & 63;   // +SHIFT mod 64
    int xx = ((wx << 3) + (tn & 7) + 4) & 63;
    tok = (size_t)b * 4096 + y * 64 + xx;
  } else {
    tok = (size_t)wr; b = (int)(wr >> 12);
  }
  const float* xp = xin + tok * 256;
  float v[8], s = 0.f, ss = 0.f;
#pragma unroll
  for (int i = 0; i < 8; ++i) { v[i] = xp[lane + (i << 5)]; s += v[i]; ss += v[i] * v[i]; }
#pragma unroll
  for (int o2 = 16; o2 >= 1; o2 >>= 1) {
    s += __shfl_xor(s, o2, 32); ss += __shfl_xor(ss, o2, 32);
  }
  float mu = s * (1.f / 256.f);
  float rstd = rsqrtf(ss * (1.f / 256.f) - mu * mu + 1e-6f);
  const float* m = mod + b * 512;
  bf16* op = out + (size_t)wr * 256;
#pragma unroll
  for (int i = 0; i < 8; ++i) {
    int c = lane + (i << 5);
    float th = (v[i] - mu) * rstd;
    op[c] = (bf16)(gamma[c] * (1.f + m[c]) * th + (beta[c] + m[256 + c]));
  }
}

// ---------------------------------------------------------------------------
// Unified GEMM: one block per 16-row M panel. The A panel (16 x K bf16) is
// staged to LDS with CDNA5 async-to-LDS loads (ASYNCcnt), then each of the 8
// waves computes a 16 x (NT*16) strip: one LDS A-fragment feeds NT WMMAs.
// MODE: 0=qkv(bias->bf16) 1=proj(residual+unwindow->f32) 2=fc1(GELU->bf16)
//       3=fc2(residual->f32)
// ---------------------------------------------------------------------------
template <int K, int NT, int MODE>
__global__ __launch_bounds__(256) void k_gemm(const bf16* __restrict__ A,
                                              const bf16* __restrict__ Bt,
                                              const float* __restrict__ bias,
                                              const float* __restrict__ aux,
                                              void* __restrict__ Cv) {
  __shared__ __align__(16) bf16 As[16 * K];
  const int N = NT * 128;
  int m0 = (int)blockIdx.x << 4;

  // ---- async stage of A panel: global -> LDS, 16B per lane per issue ----
  {
    unsigned lbase = (unsigned)(size_t)&As[0];
    const char* gbase = (const char*)(A + (size_t)m0 * K);
#pragma unroll
    for (int it = 0; it < (16 * K * 2) / 4096; ++it) {
      unsigned off = (unsigned)(it * 4096 + threadIdx.x * 16);
      unsigned la = lbase + off;
      unsigned long long ga = (unsigned long long)(gbase + off);
      asm volatile("global_load_async_to_lds_b128 %0, %1, off"
                   :: "v"(la), "v"(ga) : "memory");
    }
    asm volatile("s_wait_asynccnt 0" ::: "memory");
    __syncthreads();
  }

  const int lane = LANE;
  const int nbase = WAVE * (NT * 16);
  v8f zero = {};
  v8f acc[NT];
#pragma unroll
  for (int i = 0; i < NT; ++i) acc[i] = zero;

  const bf16* arow = &As[(lane & 15) * K + ((lane >> 4) << 4)];
  for (int k0 = 0; k0 < K; k0 += 32) {
    const bf16* pa = arow + k0;
    v16bf af = make_frag(*(const v8bf*)pa, *(const v8bf*)(pa + 8));
    __builtin_prefetch(Bt + (size_t)(nbase + (lane & 15)) * K + k0 + 32, 0, 1);
#pragma unroll
    for (int nt = 0; nt < NT; ++nt) {
      v16bf bfr = load_b_frag(Bt, K, nbase + nt * 16, k0);
      acc[nt] = wmma_bf16(af, bfr, acc[nt]);
    }
  }

  // ---- epilogues ----
  const int n_lo = lane & 15;
  const int mb = m0 + ((lane >> 4) << 3);
  if constexpr (MODE == 0) {               // qkv: +bias -> bf16
    bf16* C = (bf16*)Cv;
#pragma unroll
    for (int nt = 0; nt < NT; ++nt) {
      int n = nbase + nt * 16 + n_lo;
      float bv = bias[n];
#pragma unroll
      for (int r = 0; r < 8; ++r)
        C[(size_t)(mb + r) * N + n] = (bf16)(acc[nt][r] + bv);
    }
  } else if constexpr (MODE == 1) {        // proj: un-window + roll + residual
    float* x1 = (float*)Cv;
#pragma unroll
    for (int r = 0; r < 8; ++r) {
      int wr = mb + r;
      int w = wr >> 6, tn = wr & 63;
      int b = w >> 6, wy = (w >> 3) & 7, wx = w & 7;
      int y  = ((wy << 3) + (tn >> 3) + 4) & 63;
      int xx = ((wx << 3) + (tn & 7) + 4) & 63;
      size_t tok = (size_t)b * 4096 + y * 64 + xx;
#pragma unroll
      for (int nt = 0; nt < NT; ++nt) {
        int n = nbase + nt * 16 + n_lo;
        x1[tok * 256 + n] = aux[tok * 256 + n] + acc[nt][r] + bias[n];
      }
    }
  } else if constexpr (MODE == 2) {        // fc1: +bias, exact GELU -> bf16
    bf16* C = (bf16*)Cv;
#pragma unroll
    for (int nt = 0; nt < NT; ++nt) {
      int n = nbase + nt * 16 + n_lo;
      float bv = bias[n];
#pragma unroll
      for (int r = 0; r < 8; ++r) {
        float u = acc[nt][r] + bv;
        C[(size_t)(mb + r) * N + n] =
            (bf16)(0.5f * u * (1.f + erff(u * 0.70710678118654752f)));
      }
    }
  } else {                                 // fc2: +bias + residual -> f32
    float* C = (float*)Cv;
#pragma unroll
    for (int nt = 0; nt < NT; ++nt) {
      int n = nbase + nt * 16 + n_lo;
      float bv = bias[n];
#pragma unroll
      for (int r = 0; r < 8; ++r) {
        size_t idx = (size_t)(mb + r) * 256 + n;
        C[idx] = aux[idx] + acc[nt][r] + bv;
      }
    }
  }
}

// ---------------------------------------------------------------------------
// Windowed attention: 1 block per 8x8 window, 1 wave32 per head (NH=8, hd=32).
// ---------------------------------------------------------------------------
__global__ __launch_bounds__(256) void k_attn(const bf16* qkv, const float* biasMat,
                                              bf16* o) {
  __shared__ __align__(16) bf16 plds[8][16][64];
  int w = blockIdx.x, head = WAVE, lane = LANE;
  int lm = lane & 15, lh = lane >> 4;
  const bf16* base = qkv + (size_t)w * 64 * 768;
  const bf16* qb = base + head * 32;
  const bf16* kb = base + 256 + head * 32;
  const bf16* vb = base + 512 + head * 32;
  const float* bm = biasMat + head * 64 * 64;
  const float scale = 0.17677669529663687f;

  v16bf kf[4];
#pragma unroll
  for (int ni = 0; ni < 4; ++ni) {
    const bf16* p = kb + (size_t)((ni << 4) + lm) * 768 + (lh << 4);
    kf[ni] = make_frag(*(const v8bf*)p, *(const v8bf*)(p + 8));
  }
  v16bf vf[2][2];
#pragma unroll
  for (int kt = 0; kt < 2; ++kt)
#pragma unroll
    for (int nj = 0; nj < 2; ++nj) {
      v16bf f;
#pragma unroll
      for (int i = 0; i < 16; ++i)
        f[i] = vb[(size_t)((kt << 5) + (lh << 4) + i) * 768 + (nj << 4) + lm];
      vf[kt][nj] = f;
    }

  for (int mi = 0; mi < 4; ++mi) {
    int m0 = mi << 4;
    const bf16* pa = qb + (size_t)(m0 + lm) * 768 + (lh << 4);
    v16bf aq = make_frag(*(const v8bf*)pa, *(const v8bf*)(pa + 8));

    v8f s[4];
#pragma unroll
    for (int ni = 0; ni < 4; ++ni) {
      v8f z = {};
      s[ni] = wmma_bf16(aq, kf[ni], z);
    }

    int mbase = m0 + (lh << 3);
    float pr[4][8];
#pragma unroll
    for (int ni = 0; ni < 4; ++ni)
#pragma unroll
      for (int r = 0; r < 8; ++r)
        pr[ni][r] = s[ni][r] * scale + bm[(mbase + r) * 64 + (ni << 4) + lm];

#pragma unroll
    for (int r = 0; r < 8; ++r) {
      float mx = fmaxf(fmaxf(pr[0][r], pr[1][r]), fmaxf(pr[2][r], pr[3][r]));
#pragma unroll
      for (int o2 = 8; o2 >= 1; o2 >>= 1) mx = fmaxf(mx, __shfl_xor(mx, o2, 32));
      float sm = 0.f;
#pragma unroll
      for (int ni = 0; ni < 4; ++ni) { pr[ni][r] = __expf(pr[ni][r] - mx); sm += pr[ni][r]; }
#pragma unroll
      for (int o2 = 8; o2 >= 1; o2 >>= 1) sm += __shfl_xor(sm, o2, 32);
      float inv = 1.f / sm;
#pragma unroll
      for (int ni = 0; ni < 4; ++ni) pr[ni][r] *= inv;
    }

#pragma unroll
    for (int ni = 0; ni < 4; ++ni)
#pragma unroll
      for (int r = 0; r < 8; ++r)
        plds[head][(lh << 3) + r][(ni << 4) + lm] = (bf16)pr[ni][r];

    v8f oa0 = {}, oa1 = {};
#pragma unroll
    for (int kt = 0; kt < 2; ++kt) {
      const bf16* pp = &plds[head][lm][(kt << 5) + (lh << 4)];
      v16bf ap = make_frag(*(const v8bf*)pp, *(const v8bf*)(pp + 8));
      oa0 = wmma_bf16(ap, vf[kt][0], oa0);
      oa1 = wmma_bf16(ap, vf[kt][1], oa1);
    }
#pragma unroll
    for (int r = 0; r < 8; ++r) {
      size_t row = (size_t)((w << 6) + mbase + r) * 256 + (head << 5);
      o[row + lm]      = (bf16)oa0[r];
      o[row + 16 + lm] = (bf16)oa1[r];
    }
  }
}

// ---------------------------------------------------------------------------
// Host launcher
// ---------------------------------------------------------------------------
extern "C" void kernel_launch(void* const* d_in, const int* in_sizes, int n_in,
                              void* d_out, int out_size, void* d_ws, size_t ws_size,
                              hipStream_t stream) {
  (void)in_sizes; (void)n_in; (void)out_size; (void)ws_size;
  const float* x      = (const float*)d_in[0];
  const float* cond   = (const float*)d_in[1];
  const float* gamma1 = (const float*)d_in[4];
  const float* beta1  = (const float*)d_in[5];
  const float* mod1_w = (const float*)d_in[6];
  const float* mod1_b = (const float*)d_in[7];
  const float* qkv_w  = (const float*)d_in[8];
  const float* qkv_b  = (const float*)d_in[9];
  const float* rpb    = (const float*)d_in[10];
  const float* proj_w = (const float*)d_in[11];
  const float* proj_b = (const float*)d_in[12];
  const float* gamma2 = (const float*)d_in[13];
  const float* beta2  = (const float*)d_in[14];
  const float* mod2_w = (const float*)d_in[15];
  const float* mod2_b = (const float*)d_in[16];
  const float* fc1_w  = (const float*)d_in[17];
  const float* fc1_b  = (const float*)d_in[18];
  const float* fc2_w  = (const float*)d_in[19];
  const float* fc2_b  = (const float*)d_in[20];
  float* out = (float*)d_out;

  char* ws = (char*)d_ws;
  auto alloc = [&](size_t bytes) {
    char* p = ws; ws += (bytes + 255) & ~(size_t)255; return p;
  };
  const size_t M = 65536;
  bf16*  h1      = (bf16*)alloc(M * 256 * 2);
  bf16*  qkv     = (bf16*)alloc(M * 768 * 2);
  bf16*  obuf    = (bf16*)alloc(M * 256 * 2);
  float* x1      = (float*)alloc(M * 256 * 4);
  bf16*  h2      = (bf16*)alloc(M * 256 * 2);
  bf16*  hid     = (bf16*)alloc(M * 1024 * 2);
  bf16*  qkv_wT  = (bf16*)alloc(768 * 256 * 2);
  bf16*  proj_wT = (bf16*)alloc(256 * 256 * 2);
  bf16*  fc1_wT  = (bf16*)alloc(1024 * 256 * 2);
  bf16*  fc2_wT  = (bf16*)alloc(256 * 1024 * 2);
  float* mod1    = (float*)alloc(16 * 512 * 4);
  float* mod2    = (float*)alloc(16 * 512 * 4);
  float* biasMat = (float*)alloc(8 * 64 * 64 * 4);

  k_transpose_bf16<<<(256 * 768 + 255) / 256, 256, 0, stream>>>(qkv_w, qkv_wT, 256, 768);
  k_transpose_bf16<<<(256 * 256 + 255) / 256, 256, 0, stream>>>(proj_w, proj_wT, 256, 256);
  k_transpose_bf16<<<(256 * 1024 + 255) / 256, 256, 0, stream>>>(fc1_w, fc1_wT, 256, 1024);
  k_transpose_bf16<<<(1024 * 256 + 255) / 256, 256, 0, stream>>>(fc2_w, fc2_wT, 1024, 256);
  k_mod<<<32, 256, 0, stream>>>(cond, mod1_w, mod1_b, mod1);
  k_mod<<<32, 256, 0, stream>>>(cond, mod2_w, mod2_b, mod2);
  k_bias<<<128, 256, 0, stream>>>(rpb, biasMat);

  // adaLN1 + shifted-window partition -> bf16
  k_adaln<<<8192, 256, 0, stream>>>(x, mod1, gamma1, beta1, h1, 1);
  // QKV: (65536x256)@(256x768)
  k_gemm<256, 6, 0><<<4096, 256, 0, stream>>>(h1, qkv_wT, qkv_b, nullptr, qkv);
  // windowed attention
  k_attn<<<1024, 256, 0, stream>>>(qkv, biasMat, obuf);
  // proj + un-window + residual -> x1
  k_gemm<256, 2, 1><<<4096, 256, 0, stream>>>(obuf, proj_wT, proj_b, x, x1);
  // adaLN2
  k_adaln<<<8192, 256, 0, stream>>>(x1, mod2, gamma2, beta2, h2, 0);
  // MLP
  k_gemm<256, 8, 2><<<4096, 256, 0, stream>>>(h2, fc1_wT, fc1_b, nullptr, hid);
  k_gemm<1024, 2, 3><<<4096, 256, 0, stream>>>(hid, fc2_wT, fc2_b, x1, out);
}
